// model_vgg16_all_v1_75153337745898
// MI455X (gfx1250) — compile-verified, matrix operand order checked by reading, not run on hardware
//
#include <hip/hip_runtime.h>
#include <hip/hip_bf16.h>
#include <cstddef>
#include <cstdint>

// ---------------- CDNA5 WMMA plumbing (wave32) ----------------
typedef __attribute__((ext_vector_type(16))) __bf16 v16bf;
typedef __attribute__((ext_vector_type(8)))  float  v8f;

#define WAVES 8        // 8 wave32 => 256 threads/block
#define AROW  36       // padded LDS row stride (floats): 144B, 16B-aligned, bank-spread

static __device__ __forceinline__ __bf16 f2bf(float f) {
  // round-to-nearest-even fp32 -> bf16, pure bit math (no cvt dependency)
  unsigned u = __builtin_bit_cast(unsigned, f);
  unsigned r = u + 0x7FFFu + ((u >> 16) & 1u);
  unsigned short h = (unsigned short)(r >> 16);
  return __builtin_bit_cast(__bf16, h);
}

// CDNA5 async copy: global -> LDS, 8 bytes per lane, tracked by ASYNCcnt.
static __device__ __forceinline__ void async_copy_b64(unsigned ldsAddr, const float* g) {
  asm volatile("global_load_async_to_lds_b64 %0, %1, off"
               :: "v"(ldsAddr), "v"((unsigned long long)(uintptr_t)g)
               : "memory");
}
static __device__ __forceinline__ void wait_async_le1() {
  asm volatile("s_wait_asynccnt 0x1" ::: "memory");
}
static __device__ __forceinline__ void wait_async_0() {
  asm volatile("s_wait_asynccnt 0x0" ::: "memory");
}

// Read one lane's 16-element A fragment from an LDS tile row via 4x b128 loads.
// Element j holds K-offset halfBase + j + (j&8): two contiguous 8-float runs.
static __device__ __forceinline__ void load_afrag(const float* __restrict__ rowp,
                                                  int halfBase, v16bf& av) {
  const float4 f0 = *(const float4*)(rowp + halfBase);
  const float4 f1 = *(const float4*)(rowp + halfBase + 4);
  const float4 f2 = *(const float4*)(rowp + halfBase + 16);
  const float4 f3 = *(const float4*)(rowp + halfBase + 20);
  av[0]  = f2bf(f0.x); av[1]  = f2bf(f0.y); av[2]  = f2bf(f0.z); av[3]  = f2bf(f0.w);
  av[4]  = f2bf(f1.x); av[5]  = f2bf(f1.y); av[6]  = f2bf(f1.z); av[7]  = f2bf(f1.w);
  av[8]  = f2bf(f2.x); av[9]  = f2bf(f2.y); av[10] = f2bf(f2.z); av[11] = f2bf(f2.w);
  av[12] = f2bf(f3.x); av[13] = f2bf(f3.y); av[14] = f2bf(f3.z); av[15] = f2bf(f3.w);
}

// Per-lane B (patch) fragment gather for convolution.
template<int KS>
static __device__ __forceinline__ v16bf conv_bfrag(const float* __restrict__ X,
                                                   int kb32, int halfBase, int K, int Cin,
                                                   int b, int oh, int ow, int H, int Wd,
                                                   bool vn) {
  const int PAD = KS / 2;
  v16bf bv;
#pragma unroll
  for (int j = 0; j < 16; ++j) {
    const int kk  = kb32 + halfBase + j + (j & 8);
    const int ci  = kk / (KS * KS);
    const int rem = kk - ci * (KS * KS);
    const int ky  = rem / KS;
    const int kx  = rem - ky * KS;
    const int ih  = oh + ky - PAD;
    const int iw  = ow + kx - PAD;
    float v = 0.f;
    if (kk < K && vn && (unsigned)ih < (unsigned)H && (unsigned)iw < (unsigned)Wd)
      v = X[((b * Cin + ci) * H + ih) * Wd + iw];
    bv[j] = f2bf(v);
  }
  return bv;
}

// ---------------- Implicit-GEMM conv (KS = 1 or 3, pad = KS/2) ----------------
// D = W(Cout x K) * Patches(K x B*H*W); A tile staged to LDS (async, double-buffered).
template<int KS, bool RELU>
__global__ void conv_wmma(const float* __restrict__ X, const float* __restrict__ Wt,
                          const float* __restrict__ bias, float* __restrict__ Y,
                          int Bn, int Cin, int Cout, int H, int Wd) {
  __shared__ float As[2][16 * AROW];
  const int tid  = threadIdx.x;
  const int lane = tid & 31;
  const int wv   = tid >> 5;
  const int half = lane >> 4;
  const int hb   = half * 8;
  const int lr   = lane & 15;
  const int m0   = blockIdx.y * 16;
  const int Ntot = Bn * H * Wd;
  const int n    = (blockIdx.x * WAVES + wv) * 16 + lr;
  const bool vn  = n < Ntot;
  const int nn   = vn ? n : 0;
  const int hw   = nn % (H * Wd);
  const int b    = nn / (H * Wd);
  const int oh   = hw / Wd;
  const int ow   = hw % Wd;
  const int K    = Cin * KS * KS;
  const int nch  = (K + 31) >> 5;

  // staging map: 16 threads per tile row, 2 consecutive floats (b64) per thread
  const int srow = tid >> 4;
  const int sko  = (tid & 15) * 2;
  const unsigned lds0 = (unsigned)(uintptr_t)&As[0][srow * AROW + sko];
  const unsigned lds1 = (unsigned)(uintptr_t)&As[1][srow * AROW + sko];
  const float* wrow = Wt + (m0 + srow) * K + sko;   // A(m,kk) = Wt[m*K + kk] (contiguous)

  v8f acc = {};
  const bool fastA = ((K & 31) == 0) && (m0 + 16 <= Cout);
  if (fastA) {
    async_copy_b64(lds0, wrow);                     // prologue: chunk 0 -> buf 0
    for (int kc = 0; kc < nch; ++kc) {
      if (kc + 1 < nch) {                           // overlap next copy with this WMMA
        async_copy_b64(((kc + 1) & 1) ? lds1 : lds0, wrow + (kc + 1) * 32);
        wait_async_le1();
      } else {
        wait_async_0();
      }
      __syncthreads();
      v16bf av; load_afrag(&As[kc & 1][lr * AROW], hb, av);
      v16bf bv = conv_bfrag<KS>(X, kc * 32, hb, K, Cin, b, oh, ow, H, Wd, vn);
      acc = __builtin_amdgcn_wmma_f32_16x16x32_bf16(false, av, false, bv,
                                                    (short)0, acc, false, false);
      __syncthreads();
    }
  } else {
    for (int kc = 0; kc < nch; ++kc) {
      for (int s = tid; s < 512; s += 256) {        // guarded scalar staging (ragged tiles)
        const int row = s >> 5, ko = s & 31;
        const int m = m0 + row, kk = kc * 32 + ko;
        As[0][row * AROW + ko] = (m < Cout && kk < K) ? Wt[m * K + kk] : 0.f;
      }
      __syncthreads();
      v16bf av; load_afrag(&As[0][lr * AROW], hb, av);
      v16bf bv = conv_bfrag<KS>(X, kc * 32, hb, K, Cin, b, oh, ow, H, Wd, vn);
      acc = __builtin_amdgcn_wmma_f32_16x16x32_bf16(false, av, false, bv,
                                                    (short)0, acc, false, false);
      __syncthreads();
    }
  }
#pragma unroll
  for (int r = 0; r < 8; ++r) {
    const int m = m0 + r + half * 8;
    if (vn && m < Cout) {
      float v = acc[r] + bias[m];
      if (RELU) v = v > 0.f ? v : 0.f;
      Y[((b * Cout + m) * H + oh) * Wd + ow] = v;
    }
  }
}

// ---------------- Transposed conv 4x4 stride2 (lhs_dilation=2, pad=2), fused ReLU ----
// Parity class (oh&1, ow&1) in blockIdx.z => exactly a 2x2 tap subset; K = Cin*4.
// Weight taps not K-contiguous => scalar LDS staging for the shared A tile.
__global__ void deconv_wmma(const float* __restrict__ X, const float* __restrict__ Wt,
                            const float* __restrict__ bias, float* __restrict__ Y,
                            int Bn, int Cin, int Cout, int H, int Wd) {
  __shared__ float As[16 * AROW];
  const int tid  = threadIdx.x;
  const int lane = tid & 31;
  const int wv   = tid >> 5;
  const int half = lane >> 4;
  const int hb   = half * 8;
  const int lr   = lane & 15;
  const int po   = blockIdx.z >> 1, pw = blockIdx.z & 1;
  const int m0   = blockIdx.y * 16;
  const int Ntot = Bn * H * Wd;          // per-parity output grid is H x W
  const int n    = (blockIdx.x * WAVES + wv) * 16 + lr;
  const bool vn  = n < Ntot;
  const int nn   = vn ? n : 0;
  const int hw   = nn % (H * Wd);
  const int b    = nn / (H * Wd);
  const int ohh  = hw / Wd;
  const int oww  = hw % Wd;
  const int K    = Cin * 4;
  const int nch  = (K + 31) >> 5;

  v8f acc = {};
  for (int kc = 0; kc < nch; ++kc) {
    for (int s = tid; s < 512; s += 256) {
      const int row = s >> 5, ko = s & 31;
      const int m = m0 + row, kk = kc * 32 + ko;
      float v = 0.f;
      if (m < Cout && kk < K) {
        const int ci = kk >> 2;
        const int t  = kk & 3;
        const int ty = t >> 1, tx = t & 1;
        v = Wt[(m * Cin + ci) * 16 + (po + 2 * ty) * 4 + (pw + 2 * tx)];
      }
      As[row * AROW + ko] = v;
    }
    __syncthreads();
    v16bf av; load_afrag(&As[lr * AROW], hb, av);
    v16bf bv;
#pragma unroll
    for (int j = 0; j < 16; ++j) {
      const int kk = kc * 32 + hb + j + (j & 8);
      const int ci = kk >> 2;
      const int t  = kk & 3;
      const int ty = t >> 1, tx = t & 1;
      const int ih = ohh + po + ty - 1;   // ih = (oh + ky - 2)/2 with oh = 2*ohh+po
      const int iw = oww + pw + tx - 1;
      float v = 0.f;
      if (kk < K && vn && (unsigned)ih < (unsigned)H && (unsigned)iw < (unsigned)Wd)
        v = X[((b * Cin + ci) * H + ih) * Wd + iw];
      bv[j] = f2bf(v);
    }
    acc = __builtin_amdgcn_wmma_f32_16x16x32_bf16(false, av, false, bv,
                                                  (short)0, acc, false, false);
    __syncthreads();
  }
  const int H2 = H * 2, W2 = Wd * 2;
  const int oh = ohh * 2 + po, ow = oww * 2 + pw;
#pragma unroll
  for (int r = 0; r < 8; ++r) {
    const int m = m0 + r + half * 8;
    if (vn && m < Cout) {
      float v = acc[r] + bias[m];
      v = v > 0.f ? v : 0.f;
      Y[((b * Cout + m) * H2 + oh) * W2 + ow] = v;
    }
  }
}

// ---------------- Generic GEMM: Y[M,N] = X[M,K] * Wt[K,N] (+bias, opt ReLU) --------
// A tile (rows of X) staged to LDS: async fast path when dense, guarded scalar else.
template<bool RELU>
__global__ void gemm_wmma(const float* __restrict__ X, const float* __restrict__ Wt,
                          const float* __restrict__ bias, float* __restrict__ Y,
                          int M, int K, int N, int ldy) {
  __shared__ float As[16 * AROW];
  const int tid  = threadIdx.x;
  const int lane = tid & 31;
  const int wv   = tid >> 5;
  const int half = lane >> 4;
  const int hb   = half * 8;
  const int lr   = lane & 15;
  const int m0   = blockIdx.y * 16;
  const int col  = (blockIdx.x * WAVES + wv) * 16 + lr;
  const int nch  = (K + 31) >> 5;

  const int srow = tid >> 4;
  const int sko  = (tid & 15) * 2;
  const unsigned lds0 = (unsigned)(uintptr_t)&As[srow * AROW + sko];
  const float* xrow = X + (m0 + srow) * K + sko;
  const bool fastA = ((K & 31) == 0) && (m0 + 16 <= M);

  v8f acc = {};
  for (int kc = 0; kc < nch; ++kc) {
    if (fastA) {
      async_copy_b64(lds0, xrow + kc * 32);
      wait_async_0();
    } else {
      for (int s = tid; s < 512; s += 256) {
        const int row = s >> 5, ko = s & 31;
        const int m = m0 + row, kk = kc * 32 + ko;
        As[row * AROW + ko] = (m < M && kk < K) ? X[m * K + kk] : 0.f;
      }
    }
    __syncthreads();
    v16bf av; load_afrag(&As[lr * AROW], hb, av);
    v16bf bv;
#pragma unroll
    for (int j = 0; j < 16; ++j) {
      const int kk = kc * 32 + hb + j + (j & 8);
      float v = (kk < K && col < N) ? Wt[kk * N + col] : 0.f;  // coalesced across lanes
      bv[j] = f2bf(v);
    }
    acc = __builtin_amdgcn_wmma_f32_16x16x32_bf16(false, av, false, bv,
                                                  (short)0, acc, false, false);
    __syncthreads();
  }
#pragma unroll
  for (int r = 0; r < 8; ++r) {
    const int row = m0 + r + half * 8;
    if (row < M && col < N) {
      float v = acc[r];
      if (bias) v += bias[col];
      if (RELU) v = v > 0.f ? v : 0.f;
      Y[row * ldy + col] = v;
    }
  }
}

// ---------------- Small helper kernels ----------------
__global__ void maxpool2(const float* __restrict__ X, float* __restrict__ Y,
                         int total, int H, int Wd) {
  int t = blockIdx.x * blockDim.x + threadIdx.x;
  if (t >= total) return;
  const int OW = Wd >> 1, OH = H >> 1;
  int ow = t % OW;
  int tmp = t / OW;
  int oh = tmp % OH;
  tmp /= OH;                 // tmp = b*C + c
  const float* p = X + ((size_t)tmp * H + oh * 2) * Wd + ow * 2;
  Y[t] = fmaxf(fmaxf(p[0], p[1]), fmaxf(p[Wd], p[Wd + 1]));
}

__global__ void argmax_lm(const float* __restrict__ map, int* __restrict__ out,
                          int HW, int Wd) {
  __shared__ float sv[256];
  __shared__ int   si[256];
  const int bc = blockIdx.x;                 // b*8 + c, 64 blocks
  const float* p = map + (size_t)bc * HW;
  float bv = -3.402823466e38f; int bi = 0;
  for (int i = threadIdx.x; i < HW; i += blockDim.x) {
    float v = p[i];
    if (v > bv) { bv = v; bi = i; }
  }
  sv[threadIdx.x] = bv; si[threadIdx.x] = bi;
  __syncthreads();
  for (int s = 128; s > 0; s >>= 1) {
    if (threadIdx.x < s) {
      if (sv[threadIdx.x + s] > sv[threadIdx.x] ||
          (sv[threadIdx.x + s] == sv[threadIdx.x] && si[threadIdx.x + s] < si[threadIdx.x])) {
        sv[threadIdx.x] = sv[threadIdx.x + s];
        si[threadIdx.x] = si[threadIdx.x + s];
      }
    }
    __syncthreads();
  }
  if (threadIdx.x == 0) {
    out[bc * 2 + 0] = si[0] % Wd;   // x = idx % W
    out[bc * 2 + 1] = si[0] / Wd;   // y = idx / W
  }
}

__global__ void gather_lm(const float* __restrict__ pooled5, const int* __restrict__ lm,
                          float* __restrict__ x1) {
  int t = blockIdx.x * blockDim.x + threadIdx.x;
  if (t >= 8 * 8 * 512) return;
  const int c = t & 511;
  const int l = (t >> 9) & 7;
  const int b = t >> 12;
  const int* L = lm + (b * 8 + l) * 4;
  const int xi = (int)((float)L[2] * (7.0f / 224.0f));
  const int yi = (int)((float)L[3] * (7.0f / 224.0f));
  float v = 0.f;
  if (L[1] == 1) v = pooled5[((b * 512 + c) * 7 + xi) * 7 + yi];
  x1[t] = v;
}

// out[b, n, f] = sum_m adj[n,m] * t[(b*8+m), f] + bias[f] ; strided store
__global__ void adj_gc(const float* __restrict__ adj, const float* __restrict__ t,
                       const float* __restrict__ bias, float* __restrict__ out,
                       int F, int osB, int osN, int relu) {
  int i = blockIdx.x * blockDim.x + threadIdx.x;
  if (i >= 8 * 8 * F) return;
  const int f = i % F;
  const int n = (i / F) % 8;
  const int b = i / (F * 8);
  float acc = bias[f];
  for (int m = 0; m < 8; ++m) acc += adj[n * 8 + m] * t[(b * 8 + m) * F + f];
  if (relu) acc = acc > 0.f ? acc : 0.f;
  out[b * osB + n * osN + f] = acc;
}

// ---------------- Host-side launch helpers ----------------
static inline dim3 conv_grid(int Ntot, int Cout) {
  int nt = (Ntot + 15) / 16;
  return dim3((nt + WAVES - 1) / WAVES, (Cout + 15) / 16);
}

extern "C" void kernel_launch(void* const* d_in, const int* in_sizes, int n_in,
                              void* d_out, int out_size, void* d_ws, size_t ws_size,
                              hipStream_t stream) {
  (void)in_sizes; (void)n_in; (void)out_size; (void)ws_size;
  const float* x         = (const float*)d_in[0];
  const int*   landmarks = (const int*)d_in[1];
  const float* adjm      = (const float*)d_in[2];
  auto P = [&](int i) { return (const float*)d_in[i]; };

  // ---- workspace carve (floats) ----
  float* ws      = (float*)d_ws;
  float* bufA    = ws;                         // 25,690,112 (8x64x224x224)
  float* bufB    = bufA + 25690112;            // 25,690,112
  float* conv43  = bufB + 25690112;            // 3,211,264 (8x512x28x28)
  float* pooled5 = conv43 + 3211264;           // 200,704   (8x512x7x7)
  float* x1      = pooled5 + 200704;           // 32,768    (8x8x512)
  float* t1      = x1 + 32768;                 // 1,024     (64x16)
  float* x1g     = t1 + 1024;                  // 1,024     (8x8x16)
  float* t2      = x1g + 1024;                 // 16,384    (64x256)
  float* xc      = t2 + 16384;                 // 32,768    (8x4096)
  float* hc      = xc + 32768;                 // 32,768
  float* ha      = hc + 32768;                 // 32,768
  float* x2a     = ha + 32768;                 // 32,768

  float* outMap  = (float*)d_out;              // 8x8x224x224 = 3,211,264
  int*   outPos  = (int*)d_out + 3211264;      // 8x8x2 = 128 (int32 bit pattern)
  float* outCat  = (float*)d_out + 3211392;    // 8x50
  float* outAttr = (float*)d_out + 3211792;    // 8x2x1000

  const dim3 blk(256);
  auto c3r = [&](const float* in, int wi, int Ci, int Co, int H, int W, float* out) {
    conv_wmma<3, true><<<conv_grid(8 * H * W, Co), blk, 0, stream>>>(in, P(wi), P(wi + 1), out, 8, Ci, Co, H, W);
  };
  auto c1r = [&](const float* in, int wi, int Ci, int Co, int H, int W, float* out) {
    conv_wmma<1, true><<<conv_grid(8 * H * W, Co), blk, 0, stream>>>(in, P(wi), P(wi + 1), out, 8, Ci, Co, H, W);
  };
  auto c1n = [&](const float* in, int wi, int Ci, int Co, int H, int W, float* out) {
    conv_wmma<1, false><<<conv_grid(8 * H * W, Co), blk, 0, stream>>>(in, P(wi), P(wi + 1), out, 8, Ci, Co, H, W);
  };
  auto dec = [&](const float* in, int wi, int Ci, int Co, int H, int W, float* out) {
    dim3 g = conv_grid(8 * H * W, Co); g.z = 4;  // 4 output-parity classes
    deconv_wmma<<<g, blk, 0, stream>>>(in, P(wi), P(wi + 1), out, 8, Ci, Co, H, W);
  };
  auto pool = [&](const float* in, int C, int H, int W, float* out) {
    int total = 8 * C * (H / 2) * (W / 2);
    maxpool2<<<(total + 255) / 256, blk, 0, stream>>>(in, out, total, H, W);
  };
  auto gemm = [&](const float* X_, int wi, const float* bias, float* Y_, int M, int K, int N, int ldy, bool relu) {
    int nt = (N + 15) / 16;
    dim3 g((nt + WAVES - 1) / WAVES, (M + 15) / 16);
    if (relu) gemm_wmma<true ><<<g, blk, 0, stream>>>(X_, P(wi), bias, Y_, M, K, N, ldy);
    else      gemm_wmma<false><<<g, blk, 0, stream>>>(X_, P(wi), bias, Y_, M, K, N, ldy);
  };

  // ---------------- VGG16 trunk ----------------
  c3r(x,    3,   3,  64, 224, 224, bufA);
  c3r(bufA, 5,  64,  64, 224, 224, bufB);
  pool(bufB, 64, 224, 224, bufA);                       // -> 112
  c3r(bufA, 7,  64, 128, 112, 112, bufB);
  c3r(bufB, 9, 128, 128, 112, 112, bufA);
  pool(bufA, 128, 112, 112, bufB);                      // -> 56
  c3r(bufB, 11, 128, 256, 56, 56, bufA);
  c3r(bufA, 13, 256, 256, 56, 56, bufB);
  c3r(bufB, 15, 256, 256, 56, 56, bufA);
  pool(bufA, 256, 56, 56, bufB);                        // -> 28
  c3r(bufB, 17, 256, 512, 28, 28, bufA);
  c3r(bufA, 19, 512, 512, 28, 28, bufB);
  c3r(bufB, 21, 512, 512, 28, 28, conv43);              // vgg9 output saved
  pool(conv43, 512, 28, 28, bufA);                      // -> 14
  c3r(bufA, 23, 512, 512, 14, 14, bufB);
  c3r(bufB, 25, 512, 512, 14, 14, bufA);
  c3r(bufA, 27, 512, 512, 14, 14, bufB);
  pool(bufB, 512, 14, 14, pooled5);                     // -> 7 (pooled5)

  // ---------------- Upsample / landmark branch ----------------
  c1r(conv43, 29, 512,  64, 28, 28, bufA);              // u0
  c3r(bufA,   31,  64,  64, 28, 28, bufB);              // u1
  c3r(bufB,   33,  64,  64, 28, 28, bufA);              // u2
  c3r(bufA,   35,  64, 128, 28, 28, bufB);              // u3
  dec(bufB,   37, 128,  64, 28, 28, bufA);              // d0 -> 56
  c3r(bufA,   39,  64,  64, 56, 56, bufB);              // u4
  c3r(bufB,   41,  64,  64, 56, 56, bufA);              // u5
  dec(bufA,   43,  64,  32, 56, 56, bufB);              // d1 -> 112
  c3r(bufB,   45,  32,  32, 112, 112, bufA);            // u6
  c3r(bufA,   47,  32,  32, 112, 112, bufB);            // u7
  dec(bufB,   49,  32,  16, 112, 112, bufA);            // d2 -> 224
  c3r(bufA,   51,  16,  16, 224, 224, bufB);            // u8
  c1n(bufB,   53,  16,   8, 224, 224, outMap);          // u9 -> lm_pos_map (output 0)

  argmax_lm<<<64, blk, 0, stream>>>(outMap, outPos, 224 * 224, 224);  // output 1

  // ---------------- Graph-conv branch ----------------
  gather_lm<<<(32768 + 255) / 256, blk, 0, stream>>>(pooled5, landmarks, x1);
  gemm(x1, 55, nullptr, t1, 64, 512, 16, 16, false);                  // x1 @ gc1_w
  adj_gc<<<(8 * 8 * 16 + 255) / 256, blk, 0, stream>>>(adjm, t1, P(56), x1g, 16, 128, 16, 1);
  gemm(x1g, 57, nullptr, t2, 64, 16, 256, 256, false);                // @ gc2_w
  adj_gc<<<(8 * 8 * 256 + 255) / 256, blk, 0, stream>>>(adjm, t2, P(58), xc, 256, 4096, 256, 0);

  // ---------------- FC branch (weights stream from HBM; M=8) ----------------
  gemm(pooled5, 59, P(60), x2a,       8, 25088, 4096, 4096, true);    // fc2a + relu
  gemm(x2a,     61, P(62), xc + 2048, 8,  4096, 2048, 4096, false);   // fc2b -> xc[:,2048:]
  gemm(xc,      63, P(64), hc,        8,  4096, 4096, 4096, true);    // cat_a + relu
  gemm(hc,      65, P(66), outCat,    8,  4096,   50,   50, false);   // category (output 2)
  gemm(xc,      67, P(68), ha,        8,  4096, 4096, 4096, true);    // att_a + relu
  gemm(ha,      69, P(70), outAttr,   8,  4096, 2000, 2000, false);   // attr (output 3)
}
